// RelationalGNN_86397562126680
// MI455X (gfx1250) — matrix operand order repaired and checked by reading, not compiled, not run on hardware
//
#include <hip/hip_runtime.h>
#include <hip/hip_bf16.h>

#define NUM_SLOTS 16
#define IN_DIM    64
#define HID       128
#define G1        256
#define R_DIM     32
#define BATCH     32768
#define N_EDGES   240

typedef __attribute__((ext_vector_type(16))) __bf16 v16bf;
typedef __attribute__((ext_vector_type(8)))  float  v8f;

struct alignas(32) BfPair { uint4 lo, hi; };
struct alignas(8)  Bf4    { __bf16 h[4]; };

static __device__ __forceinline__ v16bf make_frag(uint4 lo, uint4 hi) {
  BfPair p; p.lo = lo; p.hi = hi;
  return __builtin_bit_cast(v16bf, p);
}

// A-matrix fragment (16x32 bf16, row-major source, stride in elements).
// Lane L holds row M = L%16; elements e=0..7 -> K = k0 + (L/16)*8 + e,
// e=8..15 -> K = k0 + 16 + (L/16)*8 + (e-8).  (ISA 7.12.2, 16-bit A layout)
static __device__ __forceinline__ v16bf load_a_frag(const __bf16* base, int stride,
                                                    int row0, int k0, int lane) {
  const int m = lane & 15, hi = lane >> 4;
  const __bf16* p = base + (size_t)(row0 + m) * stride + k0 + hi * 8;
  return make_frag(*(const uint4*)p, *(const uint4*)(p + 16));
}

// B-matrix fragment (32x16 bf16) loaded from a TRANSPOSED weight buffer Wt[N][K].
// Lane L holds column N = n0 + L%16; elements e=0..7 -> K = k0 + (L/16)*8 + e,
// e=8..15 -> K = k0 + 16 + (L/16)*8 + (e-8).
static __device__ __forceinline__ v16bf load_b_frag(const __bf16* base, int stride,
                                                    int n0, int k0, int lane) {
  const int n = lane & 15, hi = lane >> 4;
  const __bf16* p = base + (size_t)(n0 + n) * stride + k0 + hi * 8;
  return make_frag(*(const uint4*)p, *(const uint4*)(p + 16));
}

// Intra-wave LDS ordering: LDS ops of one wave complete in order; waiting on
// DScnt==0 makes cross-lane LDS writes of this wave visible to its own reads.
static __device__ __forceinline__ void wave_lds_fence() {
  asm volatile("s_wait_dscnt 0" ::: "memory");
}

static __device__ __forceinline__ v8f wmma_bf16(v16bf a, v16bf b, v8f c) {
  return __builtin_amdgcn_wmma_f32_16x16x32_bf16(false, a, false, b, (short)0, c,
                                                 false, false);
}

// One GCN conv for 16 batch elements:  dst = relu(A * (src @ W) + bias)
// src: [256][K] bf16 (rows = b*16+slot), Wt: [HID][K] bf16 (transposed weight),
// dst: [256][HID] bf16. Wave wv owns output column tile nt = wv (16 cols).
template <int K>
static __device__ __forceinline__
void conv_stage(const __bf16* src, const __bf16* Wt, const float* bias,
                const float* ldsA, float* scr, __bf16* dst, int lane, int wv) {
  const int nl = lane & 15, hi = lane >> 4;
  const int nt = wv;
  const float bv = bias[nt * 16 + nl];
#pragma unroll 1
  for (int b = 0; b < 16; ++b) {
    v8f acc = {};
#pragma unroll
    for (int ks = 0; ks < K / 32; ++ks) {
      v16bf a = load_a_frag(src, K, b * 16, ks * 32, lane);
      v16bf w = load_b_frag(Wt, K, nt * 16, ks * 32, lane);
      acc = wmma_bf16(a, w, acc);
    }
    // Spill the 16x16 pre-aggregation tile to per-wave scratch (C layout:
    // element r of lane L is (M = (L/16)*8 + r, N = L%16)).
#pragma unroll
    for (int r = 0; r < 8; ++r) scr[(hi * 8 + r) * 16 + nl] = acc[r];
    wave_lds_fence();
    // out[m][n] = relu( sum_s A[m][s] * tile[s][n] + bias[n] )
#pragma unroll
    for (int r = 0; r < 8; ++r) {
      const int m = hi * 8 + r;
      float v = bv;
#pragma unroll
      for (int s = 0; s < 16; ++s) v += ldsA[m * 16 + s] * scr[s * 16 + nl];
      v = v > 0.f ? v : 0.f;
      dst[(b * 16 + m) * HID + nt * 16 + nl] = (__bf16)v;
    }
    wave_lds_fence();
  }
}

// ---- prep kernels -----------------------------------------------------------

__global__ void k_build_adj(const int* __restrict__ ei, float* __restrict__ A) {
  if (threadIdx.x == 0 && blockIdx.x == 0) {
    float deg[NUM_SLOTS];
    for (int s = 0; s < NUM_SLOTS; ++s) deg[s] = 1.0f;            // self loop
    for (int e = 0; e < N_EDGES; ++e) deg[ei[N_EDGES + e]] += 1.0f;
    float dinv[NUM_SLOTS];
    for (int s = 0; s < NUM_SLOTS; ++s) dinv[s] = rsqrtf(fmaxf(deg[s], 1e-12f));
    float Al[NUM_SLOTS * NUM_SLOTS];
    for (int i = 0; i < NUM_SLOTS * NUM_SLOTS; ++i) Al[i] = 0.f;
    for (int e = 0; e < N_EDGES; ++e) {
      const int s = ei[e], d = ei[N_EDGES + e];
      Al[d * NUM_SLOTS + s] += dinv[s] * dinv[d];
    }
    for (int s = 0; s < NUM_SLOTS; ++s) Al[s * NUM_SLOTS + s] += dinv[s] * dinv[s];
    for (int i = 0; i < NUM_SLOTS * NUM_SLOTS; ++i) A[i] = Al[i];
  }
}

// W[K][N] f32  ->  Wt[N][K] bf16
__global__ void k_prep_w(const float* __restrict__ W, __bf16* __restrict__ Wt,
                         int K, int N) {
  const int i = blockIdx.x * blockDim.x + threadIdx.x;
  if (i < K * N) {
    const int k = i / N, n = i % N;
    Wt[n * K + k] = (__bf16)W[i];
  }
}

// ---- fused network kernel: one block = 16 batch elements --------------------

__global__ __launch_bounds__(256)
void k_gnn_fused(const float* __restrict__ x, const float* __restrict__ Ag,
                 const __bf16* __restrict__ W1t, const __bf16* __restrict__ W2t,
                 const __bf16* __restrict__ Wg1t, const __bf16* __restrict__ Wg2t,
                 const __bf16* __restrict__ Wmut, const __bf16* __restrict__ Wlvt,
                 const float* __restrict__ b1, const float* __restrict__ b2,
                 const float* __restrict__ bg1, const float* __restrict__ bg2,
                 const float* __restrict__ bmu, const float* __restrict__ blv,
                 float* __restrict__ out) {
  // LDS map (149 KB): A | per-wave scratch | H1 | H2 (overlaps X) | G | G2
  __shared__ __align__(16) char smem[152576];
  float*  ldsA   = (float*)(smem);             // 16x16 f32        (1 KB)
  float*  scrAll = (float*)(smem + 1024);      // 8 x 16x16 f32    (8 KB)
  __bf16* ldsH1  = (__bf16*)(smem + 9216);     // 256x128 bf16     (64 KB)
  __bf16* ldsH2  = (__bf16*)(smem + 74752);    // 256x128 bf16     (64 KB)
  __bf16* ldsX   = (__bf16*)(smem + 74752);    // 256x64 bf16 (dead after conv1)
  __bf16* ldsG   = (__bf16*)(smem + 140288);   // 16x256 bf16      (8 KB)
  __bf16* ldsG2  = (__bf16*)(smem + 148480);   // 16x128 bf16      (4 KB)

  const int tid  = threadIdx.x;
  const int lane = tid & 31, wv = tid >> 5;
  const int nl   = lane & 15, hi4 = lane >> 4;
  const int b0   = blockIdx.x * 16;
  float* scr = scrAll + wv * 256;

  // Stage-in adjacency and x tile: B128 f32 loads -> packed bf16 LDS stores.
  if (tid < 256) ldsA[tid] = Ag[tid];
  {
    const float4* xv = (const float4*)(x + (size_t)b0 * (NUM_SLOTS * IN_DIM));
#pragma unroll
    for (int i = tid; i < 256 * IN_DIM / 4; i += 256) {
      const float4 f = xv[i];
      Bf4 p;
      p.h[0] = (__bf16)f.x; p.h[1] = (__bf16)f.y;
      p.h[2] = (__bf16)f.z; p.h[3] = (__bf16)f.w;
      *(Bf4*)(ldsX + i * 4) = p;
    }
  }
  __syncthreads();

  // conv1: H1 = relu(A (X W1) + b1)    [M=16 slots, K=64, N=128]
  conv_stage<IN_DIM>(ldsX, W1t, b1, ldsA, scr, ldsH1, lane, wv);
  __syncthreads();
  // conv2: H2 = relu(A (H1 W2) + b2)   [K=128, N=128]
  conv_stage<HID>(ldsH1, W2t, b2, ldsA, scr, ldsH2, lane, wv);
  __syncthreads();

  // Head stage 1: G = relu(flat[16x2048] @ Wg1 + bg1)
  // Wave wv owns n-tiles 2*wv and 2*wv+1: one shared A-frag load per k-step,
  // two independent WMMA chains for latency hiding.
  {
    const int nt0 = wv * 2, nt1 = wv * 2 + 1;
    v8f acc0 = {}, acc1 = {};
#pragma unroll 1
    for (int ks = 0; ks < 2048 / 32; ++ks) {
      v16bf a  = load_a_frag(ldsH2, 2048, 0, ks * 32, lane);
      v16bf w0 = load_b_frag(Wg1t, 2048, nt0 * 16, ks * 32, lane);
      v16bf w1 = load_b_frag(Wg1t, 2048, nt1 * 16, ks * 32, lane);
      acc0 = wmma_bf16(a, w0, acc0);
      acc1 = wmma_bf16(a, w1, acc1);
    }
    const float bv0 = bg1[nt0 * 16 + nl];
    const float bv1 = bg1[nt1 * 16 + nl];
#pragma unroll
    for (int r = 0; r < 8; ++r) {
      float v0 = acc0[r] + bv0;
      float v1 = acc1[r] + bv1;
      v0 = v0 > 0.f ? v0 : 0.f;
      v1 = v1 > 0.f ? v1 : 0.f;
      ldsG[(hi4 * 8 + r) * G1 + nt0 * 16 + nl] = (__bf16)v0;
      ldsG[(hi4 * 8 + r) * G1 + nt1 * 16 + nl] = (__bf16)v1;
    }
  }
  __syncthreads();

  // Head stage 2: G2 = G @ Wg2 + bg2   (no relu)  [K=256, N=128]
  {
    const int nt = wv;
    v8f acc = {};
#pragma unroll
    for (int ks = 0; ks < G1 / 32; ++ks) {
      v16bf a = load_a_frag(ldsG, G1, 0, ks * 32, lane);
      v16bf w = load_b_frag(Wg2t, G1, nt * 16, ks * 32, lane);
      acc = wmma_bf16(a, w, acc);
    }
    const float bv = bg2[nt * 16 + nl];
#pragma unroll
    for (int r = 0; r < 8; ++r)
      ldsG2[(hi4 * 8 + r) * HID + nt * 16 + nl] = (__bf16)(acc[r] + bv);
  }
  __syncthreads();

  // Head stage 3: mu = G2 @ Wmu + bmu ; lv = G2 @ Wlv + blv   [K=128, N=32 each]
  if (wv < 4) {                       // wave-uniform branch: EXEC stays all-1s
    const int sel = wv >> 1, nt = wv & 1;
    const __bf16* Wt  = sel ? Wlvt : Wmut;
    const float* bias = sel ? blv : bmu;
    v8f acc = {};
#pragma unroll
    for (int ks = 0; ks < HID / 32; ++ks) {
      v16bf a = load_a_frag(ldsG2, HID, 0, ks * 32, lane);
      v16bf w = load_b_frag(Wt, HID, nt * 16, ks * 32, lane);
      acc = wmma_bf16(a, w, acc);
    }
    const float bv = bias[nt * 16 + nl];
    float* outp = out + (size_t)sel * BATCH * R_DIM;
#pragma unroll
    for (int r = 0; r < 8; ++r) {
      const int m = hi4 * 8 + r;
      outp[(size_t)(b0 + m) * R_DIM + nt * 16 + nl] = acc[r] + bv;
    }
  }
}

// ---- launcher ---------------------------------------------------------------

extern "C" void kernel_launch(void* const* d_in, const int* in_sizes, int n_in,
                              void* d_out, int out_size, void* d_ws, size_t ws_size,
                              hipStream_t stream) {
  const float* x   = (const float*)d_in[0];
  const int*   ei  = (const int*)d_in[1];
  const float* W1  = (const float*)d_in[2];
  const float* b1  = (const float*)d_in[3];
  const float* W2  = (const float*)d_in[4];
  const float* b2  = (const float*)d_in[5];
  const float* Wg1 = (const float*)d_in[6];
  const float* bg1 = (const float*)d_in[7];
  const float* Wg2 = (const float*)d_in[8];
  const float* bg2 = (const float*)d_in[9];
  const float* Wmu = (const float*)d_in[10];
  const float* bmu = (const float*)d_in[11];
  const float* Wlv = (const float*)d_in[12];
  const float* blv = (const float*)d_in[13];

  // Workspace: [0,1KB) adjacency f32, then transposed bf16 weights (~1.16 MB).
  float*  Aadj = (float*)d_ws;
  __bf16* wb   = (__bf16*)((char*)d_ws + 1024);
  __bf16* W1t  = wb;                       // [128][64]
  __bf16* W2t  = W1t + 64 * 128;           // [128][128]
  __bf16* Wg1t = W2t + 128 * 128;          // [256][2048]
  __bf16* Wg2t = Wg1t + 2048 * 256;        // [128][256]
  __bf16* Wmut = Wg2t + 256 * 128;         // [32][128]
  __bf16* Wlvt = Wmut + 128 * 32;          // [32][128]

  k_build_adj<<<1, 32, 0, stream>>>(ei, Aadj);
  k_prep_w<<<(64 * 128 + 255) / 256, 256, 0, stream>>>(W1, W1t, 64, 128);
  k_prep_w<<<(128 * 128 + 255) / 256, 256, 0, stream>>>(W2, W2t, 128, 128);
  k_prep_w<<<(2048 * 256 + 255) / 256, 256, 0, stream>>>(Wg1, Wg1t, 2048, 256);
  k_prep_w<<<(256 * 128 + 255) / 256, 256, 0, stream>>>(Wg2, Wg2t, 256, 128);
  k_prep_w<<<(128 * 32 + 255) / 256, 256, 0, stream>>>(Wmu, Wmut, 128, 32);
  k_prep_w<<<(128 * 32 + 255) / 256, 256, 0, stream>>>(Wlv, Wlvt, 128, 32);

  k_gnn_fused<<<BATCH / 16, 256, 0, stream>>>(x, Aadj, W1t, W2t, Wg1t, Wg2t,
                                              Wmut, Wlvt, b1, b2, bg1, bg2,
                                              bmu, blv, (float*)d_out);
}
